// Causal_model_vae_90048284328235
// MI455X (gfx1250) — matrix operand order repaired and checked by reading, not compiled, not run on hardware
//
#include <hip/hip_runtime.h>
#include <hip/hip_bf16.h>

typedef __attribute__((ext_vector_type(16))) _Float16 v16h;
typedef __attribute__((ext_vector_type(8)))  _Float16 v8h;
typedef __attribute__((ext_vector_type(8)))  float    v8f;
typedef __attribute__((ext_vector_type(4)))  float    v4f;

#define NROWS 524288
#define DX 128
#define DH 64
#define NS 8
#define LDS_STRIDE 72   // 16 rows * 72 halves, stride padded: m*36 mod 64 distinct for m=0..15

// ---- packed-weight workspace offsets (in halves) ----
#define OFF_MUW1 0        // 128x64 -> 8192
#define OFF_MUW2 8192     // 64x64  -> 4096
#define OFF_LVW1 12288    // 8192
#define OFF_LVW2 20480    // 4096
#define OFF_DW1  24576    // 8 * 64x64  -> 32768
#define OFF_DW2  57344    // 8 * 64x128 -> 65536
// total = 122880 halves = 240 KB

__device__ __forceinline__ v8f wmma16(v16h a, v16h b, v8f c) {
    return __builtin_amdgcn_wmma_f32_16x16x32_f16(false, a, false, b, (short)0, c,
                                                  false, false);
}

__device__ __forceinline__ v8f zero8() {
    v8f z = {0.f, 0.f, 0.f, 0.f, 0.f, 0.f, 0.f, 0.f};
    return z;
}

// Convert f32 weights (row-major K x Nn, batched) into f16 B-fragment-packed layout.
// Fragment f = nt*(K/32)+kb, within fragment: halves[lane*16+j] = W[k][n],
//   n = nt*16 + (lane%16), k = kb*32 + (lane/16)*8 + (j%8) + (j/8)*16   (dense f16 B layout, wave32)
__global__ void pack_w_kernel(const float* __restrict__ src, _Float16* __restrict__ dst,
                              int K, int Nn) {
    int b   = blockIdx.y;
    int per = K * Nn;
    int e   = blockIdx.x * blockDim.x + threadIdx.x;
    if (e >= per) return;
    int f    = e >> 9;        // /512
    int idx  = e & 511;
    int ln   = idx >> 4;
    int j    = idx & 15;
    int nKb  = K >> 5;
    int nt   = f / nKb;
    int kb   = f - nt * nKb;
    int n    = nt * 16 + (ln & 15);
    int k    = kb * 32 + (ln >> 4) * 8 + (j & 7) + (j >> 3) * 16;
    dst[(size_t)b * per + e] = (_Float16)src[(size_t)b * per + (size_t)k * Nn + n];
}

// Load two A fragments (K=64) from a 16x64 f16 LDS tile (stride LDS_STRIDE halves).
__device__ __forceinline__ void load_a_lds(const _Float16* hl, int m, int hv, v16h* afrag) {
    asm volatile("s_wait_dscnt 0" ::: "memory");
#pragma unroll
    for (int kb = 0; kb < 2; ++kb) {
        const v8h* plo = (const v8h*)(hl + m * LDS_STRIDE + kb * 32 + hv * 8);
        const v8h* phi = (const v8h*)(hl + m * LDS_STRIDE + kb * 32 + hv * 8 + 16);
        v8h lo = *plo;   // K = kb*32 + hv*8 + (0..7)
        v8h hi = *phi;   // K = kb*32 + hv*8 + 16 + (0..7)
        v16h a;
#pragma unroll
        for (int j = 0; j < 8; ++j) { a[j] = lo[j]; a[8 + j] = hi[j]; }
        afrag[kb] = a;
    }
}

// Encoder layer1: [16x128]@[128x64] + bias, LeakyReLU, write f16 tile to LDS.
__device__ __forceinline__ void enc_layer1(const v16h* a_x, const _Float16* wf,
                                           const float* __restrict__ b1,
                                           _Float16* hl, int lane, int m, int hv) {
#pragma unroll
    for (int nt = 0; nt < 4; ++nt) {
        v8f acc = zero8();
#pragma unroll
        for (int kb = 0; kb < 4; ++kb) {
            v16h b = ((const v16h*)(wf + (size_t)(nt * 4 + kb) * 512))[lane];
            acc = wmma16(a_x[kb], b, acc);
        }
        float bias = b1[nt * 16 + m];
#pragma unroll
        for (int r = 0; r < 8; ++r) {
            float x = acc[r] + bias;
            x = x > 0.f ? x : 0.01f * x;
            hl[(r + 8 * hv) * LDS_STRIDE + nt * 16 + m] = (_Float16)x;
        }
    }
}

__global__ __launch_bounds__(128) void vae_main_kernel(
    const float* __restrict__ data, const int* __restrict__ s,
    const float* __restrict__ eps,
    const float* __restrict__ mu_b1, const float* __restrict__ mu_b2,
    const float* __restrict__ lv_b1, const float* __restrict__ lv_b2,
    const float* __restrict__ dec_b1, const float* __restrict__ dec_b2,
    const _Float16* __restrict__ wpk, float* __restrict__ out) {
    __shared__ _Float16 lds_h[4][16 * LDS_STRIDE];

    const int lane = threadIdx.x & 31;
    const int wv   = threadIdx.x >> 5;
    const int m    = lane & 15;   // column / row-in-tile index
    const int hv   = lane >> 4;   // lane half (0/1)
    const int row0 = (blockIdx.x * 4 + wv) * 16;
    _Float16* hl = lds_h[wv];

    const _Float16* w_muW1 = wpk + OFF_MUW1;
    const _Float16* w_muW2 = wpk + OFF_MUW2;
    const _Float16* w_lvW1 = wpk + OFF_LVW1;
    const _Float16* w_lvW2 = wpk + OFF_LVW2;
    const _Float16* w_dW1  = wpk + OFF_DW1;
    const _Float16* w_dW2  = wpk + OFF_DW2;

    float* out_rc = out;
    float* out_mu = out + (size_t)NROWS * DX;
    float* out_lv = out_mu + (size_t)NROWS * DH;
    float* out_hs = out_lv + (size_t)NROWS * DH;

    // ---- load data tile into A fragments (f32 -> f16 in regs) ----
    v16h a_x[4];
    {
        const float* rp = data + (size_t)(row0 + m) * DX;
#pragma unroll
        for (int kb = 0; kb < 4; ++kb) {
            int c0 = kb * 32 + hv * 8;
            v4f f0 = *(const v4f*)(rp + c0);
            v4f f1 = *(const v4f*)(rp + c0 + 4);
            v4f f2 = *(const v4f*)(rp + c0 + 16);
            v4f f3 = *(const v4f*)(rp + c0 + 20);
            v16h a;
#pragma unroll
            for (int j = 0; j < 4; ++j) {
                a[j]      = (_Float16)f0[j];
                a[4 + j]  = (_Float16)f1[j];
                a[8 + j]  = (_Float16)f2[j];
                a[12 + j] = (_Float16)f3[j];
            }
            a_x[kb] = a;
        }
    }

    v8f mu[4], lv[4];

    // ---- mu encoder ----
    enc_layer1(a_x, w_muW1, mu_b1, hl, lane, m, hv);
    {
        v16h ah[2];
        load_a_lds(hl, m, hv, ah);
#pragma unroll
        for (int nt = 0; nt < 4; ++nt) {
            v8f c = zero8();
#pragma unroll
            for (int kb = 0; kb < 2; ++kb) {
                v16h b = ((const v16h*)(w_muW2 + (size_t)(nt * 2 + kb) * 512))[lane];
                c = wmma16(ah[kb], b, c);
            }
            float bias = mu_b2[nt * 16 + m];
#pragma unroll
            for (int r = 0; r < 8; ++r) c[r] += bias;
            mu[nt] = c;
        }
    }

    // ---- logvar encoder ----
    enc_layer1(a_x, w_lvW1, lv_b1, hl, lane, m, hv);
    {
        v16h ah[2];
        load_a_lds(hl, m, hv, ah);
#pragma unroll
        for (int nt = 0; nt < 4; ++nt) {
            v8f c = zero8();
#pragma unroll
            for (int kb = 0; kb < 2; ++kb) {
                v16h b = ((const v16h*)(w_lvW2 + (size_t)(nt * 2 + kb) * 512))[lane];
                c = wmma16(ah[kb], b, c);
            }
            float bias = lv_b2[nt * 16 + m];
#pragma unroll
            for (int r = 0; r < 8; ++r) c[r] += bias;
            lv[nt] = c;
        }
    }

    // ---- reparameterize, write mu/lv/hs, stage hs into LDS ----
#pragma unroll
    for (int nt = 0; nt < 4; ++nt) {
#pragma unroll
        for (int r = 0; r < 8; ++r) {
            int M = r + 8 * hv;
            size_t ro = (size_t)(row0 + M);
            int c = nt * 16 + m;
            float e   = eps[ro * DH + c];
            float muv = mu[nt][r];
            float lvv = lv[nt][r];
            float hsv = e * __expf(0.5f * lvv) + muv;
            out_mu[ro * DH + c] = muv;
            out_lv[ro * DH + c] = lvv;
            out_hs[ro * DH + c] = hsv;
            hl[M * LDS_STRIDE + c] = (_Float16)hsv;
        }
    }
    v16h a_h[2];
    load_a_lds(hl, m, hv, a_h);

    // ---- group ids for our 16 rows (per-lane, per C-layout row r) ----
    int su[8];
#pragma unroll
    for (int r = 0; r < 8; ++r) su[r] = s[row0 + r + 8 * hv];

    // ---- decoder: all 8 groups, per-lane select into accumulators ----
    v8f racc[8];
#pragma unroll
    for (int nt = 0; nt < 8; ++nt) racc[nt] = zero8();

    for (int g = 0; g < NS; ++g) {
        const _Float16* wg1 = w_dW1 + (size_t)g * 4096;
        const _Float16* wg2 = w_dW2 + (size_t)g * 8192;
        // layer1: [16x64]@[64x64] + bias + LeakyReLU -> LDS
#pragma unroll
        for (int nt = 0; nt < 4; ++nt) {
            v8f c = zero8();
#pragma unroll
            for (int kb = 0; kb < 2; ++kb) {
                v16h b = ((const v16h*)(wg1 + (size_t)(nt * 2 + kb) * 512))[lane];
                c = wmma16(a_h[kb], b, c);
            }
            float bias = dec_b1[g * DH + nt * 16 + m];
#pragma unroll
            for (int r = 0; r < 8; ++r) {
                float x = c[r] + bias;
                x = x > 0.f ? x : 0.01f * x;
                hl[(r + 8 * hv) * LDS_STRIDE + nt * 16 + m] = (_Float16)x;
            }
        }
        v16h ag[2];
        load_a_lds(hl, m, hv, ag);
        // layer2: [16x64]@[64x128] + bias, select rows where s==g
#pragma unroll
        for (int nt = 0; nt < 8; ++nt) {
            v8f c = zero8();
#pragma unroll
            for (int kb = 0; kb < 2; ++kb) {
                v16h b = ((const v16h*)(wg2 + (size_t)(nt * 2 + kb) * 512))[lane];
                c = wmma16(ag[kb], b, c);
            }
            float bias = dec_b2[g * DX + nt * 16 + m];
#pragma unroll
            for (int r = 0; r < 8; ++r) {
                float y = c[r] + bias;
                racc[nt][r] = (su[r] == g) ? y : racc[nt][r];
            }
        }
    }

    // ---- write reconst ----
#pragma unroll
    for (int nt = 0; nt < 8; ++nt) {
#pragma unroll
        for (int r = 0; r < 8; ++r) {
            size_t ro = (size_t)(row0 + r + 8 * hv);
            out_rc[ro * DX + nt * 16 + m] = racc[nt][r];
        }
    }
}

extern "C" void kernel_launch(void* const* d_in, const int* in_sizes, int n_in,
                              void* d_out, int out_size, void* d_ws, size_t ws_size,
                              hipStream_t stream) {
    const float* data   = (const float*)d_in[0];
    const int*   s      = (const int*)d_in[1];
    const float* eps    = (const float*)d_in[2];
    const float* mu_W1  = (const float*)d_in[3];
    const float* mu_b1  = (const float*)d_in[4];
    const float* mu_W2  = (const float*)d_in[5];
    const float* mu_b2  = (const float*)d_in[6];
    const float* lv_W1  = (const float*)d_in[7];
    const float* lv_b1  = (const float*)d_in[8];
    const float* lv_W2  = (const float*)d_in[9];
    const float* lv_b2  = (const float*)d_in[10];
    const float* dec_W1 = (const float*)d_in[11];
    const float* dec_b1 = (const float*)d_in[12];
    const float* dec_W2 = (const float*)d_in[13];
    const float* dec_b2 = (const float*)d_in[14];

    _Float16* wpk = (_Float16*)d_ws;
    float*    out = (float*)d_out;

    // Pack all weights into f16 WMMA B-fragment layout (L2-resident, ~240 KB).
    pack_w_kernel<<<dim3(32, 1), 256, 0, stream>>>(mu_W1, wpk + OFF_MUW1, 128, 64);
    pack_w_kernel<<<dim3(16, 1), 256, 0, stream>>>(mu_W2, wpk + OFF_MUW2, 64, 64);
    pack_w_kernel<<<dim3(32, 1), 256, 0, stream>>>(lv_W1, wpk + OFF_LVW1, 128, 64);
    pack_w_kernel<<<dim3(16, 1), 256, 0, stream>>>(lv_W2, wpk + OFF_LVW2, 64, 64);
    pack_w_kernel<<<dim3(16, 8), 256, 0, stream>>>(dec_W1, wpk + OFF_DW1, 64, 64);
    pack_w_kernel<<<dim3(32, 8), 256, 0, stream>>>(dec_W2, wpk + OFF_DW2, 64, 128);

    // 4 waves/block, 16 rows/wave -> 64 rows/block.
    vae_main_kernel<<<NROWS / 64, 128, 0, stream>>>(
        data, s, eps, mu_b1, mu_b2, lv_b1, lv_b2, dec_b1, dec_b2, wpk, out);
}
